// MambaController_44083544326768
// MI455X (gfx1250) — compile-verified, matrix operand order checked by reading, not run on hardware
//
#include <hip/hip_runtime.h>
#include <hip/hip_bf16.h>

#define BQ 32
#define LQ 4096
#define OBS 32
#define ACT 8
#define DMODEL 64
#define DSTATE 16
#define DINNER 128
#define DTRANK 4
#define NPROJ 160            // 128 dt-pre + 16 B + 16 C (fused x_proj/dt_proj GEMM)
#define BLTOT (BQ * LQ)      // 131072 tokens

typedef __attribute__((ext_vector_type(16))) _Float16 v16h;
typedef __attribute__((ext_vector_type(8)))  float    v8f;

// K-index mapping for 16-bit A/B fragments (ISA 7.12.2, 16-bit 16x32 layout):
// lanes 0-15: VGPR0..3 -> K=0..7, VGPR4..7 -> K=16..23 ; lanes 16-31: +8.
__device__ __forceinline__ int kmap16(int h, int hi) {
    int v = h >> 1, p = h & 1;
    int base = (v < 4) ? (2 * v + p) : (16 + 2 * (v - 4) + p);
    return base + (hi ? 8 : 0);
}

// ---------------------------------------------------------------------------
// Weight repack: fp32 row-major -> f16 WMMA B-fragment tiles.
// Also builds the fused dt-matrix  M_dt = dt_proj_W @ x_proj_W[:4,:]  and
// A = -exp(A_log).  Tiny work; one block per layer.
// ---------------------------------------------------------------------------
__global__ void prep_kernel(const float* __restrict__ in_proj_W,
                            const float* __restrict__ x_proj_W,
                            const float* __restrict__ dt_proj_W,
                            const float* __restrict__ A_log,
                            const float* __restrict__ out_proj_W,
                            _Float16* __restrict__ in_frag,
                            _Float16* __restrict__ w2_frag,
                            _Float16* __restrict__ out_frag,
                            float* __restrict__ Aneg)
{
    const int layer = blockIdx.x;
    const int tid = threadIdx.x;
    const float* Wi = in_proj_W + layer * (2 * DINNER) * DMODEL;            // 256 x 64
    const float* Xp = x_proj_W + layer * (DTRANK + 2 * DSTATE) * DINNER;    // 36 x 128
    const float* Dt = dt_proj_W + layer * DINNER * DTRANK;                  // 128 x 4
    const float* Op = out_proj_W + layer * DMODEL * DINNER;                 // 64 x 128
    _Float16* inF  = in_frag  + layer * (16 * 2 * 32 * 16);
    _Float16* w2F  = w2_frag  + layer * (10 * 4 * 32 * 16);
    _Float16* outF = out_frag + layer * (4 * 4 * 32 * 16);
    float* An = Aneg + layer * DINNER * DSTATE;

    // in_proj: B = Wi^T, N=256 (16 ntiles), K=64 (2 ktiles)
    for (int idx = tid; idx < 16 * 2 * 32 * 16; idx += blockDim.x) {
        int h = idx & 15, lane = (idx >> 4) & 31, kt = (idx >> 9) & 1, nt = idx >> 10;
        int k = kt * 32 + kmap16(h, lane >> 4);
        int n = nt * 16 + (lane & 15);
        inF[idx] = (_Float16)Wi[n * DMODEL + k];
    }
    // fused W2: rows 0..127 = M_dt, 128..143 = B rows, 144..159 = C rows; K=128
    for (int idx = tid; idx < 10 * 4 * 32 * 16; idx += blockDim.x) {
        int h = idx & 15, lane = (idx >> 4) & 31, kt = (idx >> 9) & 3, nt = idx >> 11;
        int k = kt * 32 + kmap16(h, lane >> 4);
        int n = nt * 16 + (lane & 15);
        float val;
        if (n < DINNER) {
            val = 0.f;
            #pragma unroll
            for (int r = 0; r < DTRANK; ++r)
                val += Dt[n * DTRANK + r] * Xp[r * DINNER + k];
        } else if (n < DINNER + DSTATE) {
            val = Xp[(DTRANK + (n - DINNER)) * DINNER + k];
        } else {
            val = Xp[(DTRANK + DSTATE + (n - DINNER - DSTATE)) * DINNER + k];
        }
        w2F[idx] = (_Float16)val;
    }
    // out_proj: N=64 (4 ntiles), K=128 (4 ktiles)
    for (int idx = tid; idx < 4 * 4 * 32 * 16; idx += blockDim.x) {
        int h = idx & 15, lane = (idx >> 4) & 31, kt = (idx >> 9) & 3, nt = idx >> 11;
        int k = kt * 32 + kmap16(h, lane >> 4);
        int n = nt * 16 + (lane & 15);
        outF[idx] = (_Float16)Op[n * DINNER + k];
    }
    for (int idx = tid; idx < DINNER * DSTATE; idx += blockDim.x)
        An[idx] = -__expf(A_log[layer * DINNER * DSTATE + idx]);
}

// ---------------------------------------------------------------------------
// WMMA GEMM with A-panel reuse: one wave per 16-row m-tile; A fragments for
// the whole K extent are loaded/converted ONCE into registers, then swept
// against all NT weight n-tiles (B fragments are tiny and L0/L2-resident).
// Cuts activation re-reads by NT x.  All dims divide evenly -> EXEC all ones.
// ---------------------------------------------------------------------------
template <int NT, int KT, bool ADDC>
__global__ void wmma_gemm_kernel(const float* __restrict__ A, int ldA,
                                 const _Float16* __restrict__ Bfrag,
                                 float* __restrict__ C, int ldC)
{
    const int lane = threadIdx.x & 31;
    const int mt = (blockIdx.x * blockDim.x + threadIdx.x) >> 5;   // one wave / m-tile
    const int hi = lane >> 4;
    const int m = mt * 16 + (lane & 15);

    // Load + convert the full A panel for this m-tile once.
    v16h a[KT];
    #pragma unroll
    for (int kt = 0; kt < KT; ++kt) {
        const float* arow = A + (size_t)m * ldA + kt * 32 + (hi ? 8 : 0);
        #pragma unroll
        for (int h = 0; h < 16; ++h) {
            int v = h >> 1, p = h & 1;
            int kb = (v < 4) ? (2 * v + p) : (16 + 2 * (v - 4) + p);
            a[kt][h] = (_Float16)arow[kb];
        }
    }

    const int rbase = mt * 16 + (hi ? 8 : 0);
    const int colbase = lane & 15;
    for (int nt = 0; nt < NT; ++nt) {
        v8f acc = {0.f, 0.f, 0.f, 0.f, 0.f, 0.f, 0.f, 0.f};
        #pragma unroll
        for (int kt = 0; kt < KT; ++kt) {
            v16h b = *(const v16h*)(Bfrag + (((size_t)(nt * KT + kt)) * 32 + lane) * 16);
            acc = __builtin_amdgcn_wmma_f32_16x16x32_f16(false, a[kt], false, b,
                                                         (short)0, acc, false, false);
        }
        const int col = nt * 16 + colbase;
        #pragma unroll
        for (int r = 0; r < 8; ++r) {
            size_t idx = (size_t)(rbase + r) * ldC + col;
            if (ADDC) C[idx] += acc[r]; else C[idx] = acc[r];
        }
    }
}

// ---------------------------------------------------------------------------
// obs -> normalized -> LN(32) -> W_in GEMV -> x[BL,64]
// ---------------------------------------------------------------------------
__global__ void pre_kernel(const float* __restrict__ obs,
                           const float* __restrict__ omean, const float* __restrict__ oscale,
                           const float* __restrict__ lnw, const float* __restrict__ lnb,
                           const float* __restrict__ W_in, const float* __restrict__ b_in,
                           float* __restrict__ x)
{
    int bl = blockIdx.x * blockDim.x + threadIdx.x;
    if (bl >= BLTOT) return;
    float t[OBS];
    const float* o = obs + (size_t)bl * OBS;
    float m = 0.f;
    #pragma unroll
    for (int i = 0; i < OBS; ++i) { t[i] = (o[i] - omean[i]) / oscale[i]; m += t[i]; }
    m *= (1.f / OBS);
    float var = 0.f;
    #pragma unroll
    for (int i = 0; i < OBS; ++i) { float d = t[i] - m; var += d * d; }
    float r = rsqrtf(var * (1.f / OBS) + 1e-5f);
    #pragma unroll
    for (int i = 0; i < OBS; ++i) t[i] = (t[i] - m) * r * lnw[i] + lnb[i];
    float* xo = x + (size_t)bl * DMODEL;
    for (int d = 0; d < DMODEL; ++d) {
        float acc = b_in[d];
        #pragma unroll
        for (int i = 0; i < OBS; ++i) acc += W_in[d * OBS + i] * t[i];
        xo[d] = acc;
    }
}

__global__ void ln_kernel(const float* __restrict__ x,
                          const float* __restrict__ w, const float* __restrict__ b,
                          float* __restrict__ out)
{
    int bl = blockIdx.x * blockDim.x + threadIdx.x;
    if (bl >= BLTOT) return;
    const float* xi = x + (size_t)bl * DMODEL;
    float t[DMODEL];
    float m = 0.f;
    #pragma unroll
    for (int i = 0; i < DMODEL; ++i) { t[i] = xi[i]; m += t[i]; }
    m *= (1.f / DMODEL);
    float var = 0.f;
    #pragma unroll
    for (int i = 0; i < DMODEL; ++i) { float d = t[i] - m; var += d * d; }
    float r = rsqrtf(var * (1.f / DMODEL) + 1e-5f);
    float* xo = out + (size_t)bl * DMODEL;
    #pragma unroll
    for (int i = 0; i < DMODEL; ++i) xo[i] = (t[i] - m) * r * w[i] + b[i];
}

// depthwise causal conv (k=4) + bias + silu ; u lives in xz cols [0,128)
__global__ void conv_kernel(const float* __restrict__ xz,
                            const float* __restrict__ cw, const float* __restrict__ cb,
                            float* __restrict__ uconv)
{
    int idx = blockIdx.x * blockDim.x + threadIdx.x;   // over BLTOT*DINNER
    if (idx >= BLTOT * DINNER) return;
    int d = idx & (DINNER - 1);
    int bl = idx >> 7;
    int l = bl & (LQ - 1);
    float acc = cb[d];
    #pragma unroll
    for (int j = 0; j < 4; ++j) {
        int lj = l - 3 + j;
        if (lj >= 0) acc += cw[d * 4 + j] * xz[(size_t)(bl - 3 + j) * 256 + d];
    }
    uconv[(size_t)bl * DINNER + d] = acc / (1.f + __expf(-acc));   // silu
}

// ---------------------------------------------------------------------------
// Selective scan: one thread per (b, d, s); sequential over L.
// 16-lane xor-shuffle reduction over the state dim; fp32 throughout.
// ---------------------------------------------------------------------------
__global__ void scan_kernel(const float* __restrict__ proj,
                            const float* __restrict__ uconv,
                            const float* __restrict__ xz,
                            const float* __restrict__ Aneg,
                            const float* __restrict__ Dp,
                            const float* __restrict__ dtb,
                            float* __restrict__ y)
{
    const int s = threadIdx.x & 15;
    const int dl = threadIdx.x >> 4;
    const int d = blockIdx.x * 16 + dl;
    const int b = blockIdx.y;
    const float A_ds = Aneg[d * DSTATE + s];
    const float Dd = Dp[d];
    const float dtbd = dtb[d];
    const size_t bl0 = (size_t)b * LQ;
    const float* pr = proj  + bl0 * NPROJ;
    const float* ur = uconv + bl0 * DINNER;
    const float* zr = xz    + bl0 * 256 + DINNER;   // z = xz cols [128,256)
    float* yr = y + bl0 * DINNER;
    float h = 0.f;
    for (int l = 0; l < LQ; ++l) {
        float dtin = pr[d] + dtbd;
        float Bs = pr[DINNER + s];
        float Cs = pr[DINNER + DSTATE + s];
        float uv = ur[d];
        // softplus (numerically stable)
        float dt = fmaxf(dtin, 0.f) + __logf(1.f + __expf(-fabsf(dtin)));
        h = h * __expf(dt * A_ds) + (dt * uv) * Bs;
        float part = h * Cs;
        part += __shfl_xor(part, 1, 32);
        part += __shfl_xor(part, 2, 32);
        part += __shfl_xor(part, 4, 32);
        part += __shfl_xor(part, 8, 32);
        if (s == 0) {
            float zv = zr[d];
            yr[d] = (part + uv * Dd) * (zv / (1.f + __expf(-zv)));
        }
        pr += NPROJ; ur += DINNER; zr += 256; yr += DINNER;
        __builtin_prefetch(pr + NPROJ, 0, 0);   // global_prefetch_b8
    }
}

__global__ void head_kernel(const float* __restrict__ x,
                            const float* __restrict__ W, const float* __restrict__ bo,
                            float* __restrict__ out)
{
    int bl = blockIdx.x * blockDim.x + threadIdx.x;
    if (bl >= BLTOT) return;
    const float* xi = x + (size_t)bl * DMODEL;
    float* oo = out + (size_t)bl * ACT;
    #pragma unroll
    for (int a = 0; a < ACT; ++a) {
        float acc = bo[a];
        #pragma unroll
        for (int j = 0; j < DMODEL; ++j) acc += W[a * DMODEL + j] * xi[j];
        oo[a] = acc;
    }
}

// ---------------------------------------------------------------------------
extern "C" void kernel_launch(void* const* d_in, const int* in_sizes, int n_in,
                              void* d_out, int out_size, void* d_ws, size_t ws_size,
                              hipStream_t stream)
{
    const float* obs       = (const float*)d_in[0];
    const float* obs_mean  = (const float*)d_in[1];
    const float* obs_scale = (const float*)d_in[2];
    const float* ln_in_w   = (const float*)d_in[3];
    const float* ln_in_b   = (const float*)d_in[4];
    const float* W_in      = (const float*)d_in[5];
    const float* b_in      = (const float*)d_in[6];
    const float* norm_w    = (const float*)d_in[7];
    const float* norm_b    = (const float*)d_in[8];
    const float* in_proj_W = (const float*)d_in[9];
    const float* conv_w    = (const float*)d_in[10];
    const float* conv_b    = (const float*)d_in[11];
    const float* x_proj_W  = (const float*)d_in[12];
    const float* dt_proj_W = (const float*)d_in[13];
    const float* dt_proj_b = (const float*)d_in[14];
    const float* A_log     = (const float*)d_in[15];
    const float* Dp        = (const float*)d_in[16];
    const float* out_proj_W= (const float*)d_in[17];
    const float* W_out     = (const float*)d_in[18];
    const float* b_out     = (const float*)d_in[19];

    char* ws = (char*)d_ws;
    size_t off = 0;
    auto take = [&](size_t bytes) {
        char* p = ws + off;
        off = (off + bytes + 255) & ~(size_t)255;
        return p;
    };
    float* x     = (float*)take((size_t)BLTOT * DMODEL * 4);
    float* xln   = (float*)take((size_t)BLTOT * DMODEL * 4);
    float* xz    = (float*)take((size_t)BLTOT * 256 * 4);
    float* uconv = (float*)take((size_t)BLTOT * DINNER * 4);
    float* proj  = (float*)take((size_t)BLTOT * NPROJ * 4);
    float* y     = (float*)take((size_t)BLTOT * DINNER * 4);
    _Float16* in_frag  = (_Float16*)take((size_t)2 * 16 * 2 * 32 * 16 * 2);
    _Float16* w2_frag  = (_Float16*)take((size_t)2 * 10 * 4 * 32 * 16 * 2);
    _Float16* out_frag = (_Float16*)take((size_t)2 * 4 * 4 * 32 * 16 * 2);
    float* Aneg = (float*)take((size_t)2 * DINNER * DSTATE * 4);

    // 8192 m-tiles, one wave each -> 1024 blocks of 256 threads (8 waves).
    const int gemmBlocks = (BLTOT / 16) * 32 / 256;

    prep_kernel<<<2, 256, 0, stream>>>(in_proj_W, x_proj_W, dt_proj_W, A_log,
                                       out_proj_W, in_frag, w2_frag, out_frag, Aneg);
    pre_kernel<<<BLTOT / 256, 256, 0, stream>>>(obs, obs_mean, obs_scale,
                                                ln_in_w, ln_in_b, W_in, b_in, x);
    for (int layer = 0; layer < 2; ++layer) {
        ln_kernel<<<BLTOT / 256, 256, 0, stream>>>(x, norm_w + layer * DMODEL,
                                                   norm_b + layer * DMODEL, xln);
        // in_proj: [BL,64] x [64,256] -> xz
        wmma_gemm_kernel<16, 2, false><<<gemmBlocks, 256, 0, stream>>>(
            xln, DMODEL, in_frag + (size_t)layer * 16 * 2 * 32 * 16, xz, 256);
        conv_kernel<<<BLTOT * DINNER / 256, 256, 0, stream>>>(
            xz, conv_w + layer * DINNER * 4, conv_b + layer * DINNER, uconv);
        // fused x_proj + dt_proj: [BL,128] x [128,160] -> proj (dt-pre | B | C)
        wmma_gemm_kernel<10, 4, false><<<gemmBlocks, 256, 0, stream>>>(
            uconv, DINNER, w2_frag + (size_t)layer * 10 * 4 * 32 * 16, proj, NPROJ);
        scan_kernel<<<dim3(DINNER / 16, BQ), 256, 0, stream>>>(
            proj, uconv, xz, Aneg + layer * DINNER * DSTATE,
            Dp + layer * DINNER, dt_proj_b + layer * DINNER, y);
        // out_proj with residual: x += [BL,128] x [128,64]
        wmma_gemm_kernel<4, 4, true><<<gemmBlocks, 256, 0, stream>>>(
            y, DINNER, out_frag + (size_t)layer * 4 * 4 * 32 * 16, x, DMODEL);
    }
    head_kernel<<<BLTOT / 256, 256, 0, stream>>>(x, W_out, b_out, (float*)d_out);
}
